// PerformerAttBlock_14267881357959
// MI455X (gfx1250) — compile-verified, matrix operand order checked by reading, not run on hardware
//
#include <hip/hip_runtime.h>
#include <hip/hip_bf16.h>
#include <stdint.h>

// ---- CDNA5 (gfx1250) vector types ----
typedef __attribute__((ext_vector_type(16))) __bf16       v16bf;
typedef __attribute__((ext_vector_type(8)))  float        v8f;
typedef __attribute__((ext_vector_type(4)))  float        v4f;
typedef __attribute__((ext_vector_type(4)))  unsigned int uv4;
typedef __attribute__((ext_vector_type(2)))  unsigned int uv2;

#define NB     16      // batch
#define CCH    512     // channels
#define NFD    256     // num features
#define HW     4096    // 64*64 pixels
#define GROUPS 32
#define CPG    16      // channels per group

static __device__ __forceinline__ unsigned short f2bf(float x) {
  union { float f; unsigned u; } v; v.f = x;
  unsigned r = v.u + 0x7FFFu + ((v.u >> 16) & 1u);   // round-to-nearest-even
  return (unsigned short)(r >> 16);
}

// ------------------------------------------------------------------
// GroupNorm pass 1: per (batch, group) mean & rstd (group = 16ch*4096px
// contiguous in (b,c,p) layout). Vectorized b128 loads.
// ------------------------------------------------------------------
__global__ void gn_stats_kernel(const float* __restrict__ x,
                                float* __restrict__ mu, float* __restrict__ rs) {
  __shared__ float ss[256], ss2[256];
  const int bg = blockIdx.x;
  const v4f* p4 = (const v4f*)(x + (size_t)bg * (CPG * HW));
  float s = 0.f, s2 = 0.f;
  for (int i = threadIdx.x; i < (CPG * HW) / 4; i += 256) {
    v4f v = p4[i];
    s  += v[0] + v[1] + v[2] + v[3];
    s2 += v[0]*v[0] + v[1]*v[1] + v[2]*v[2] + v[3]*v[3];
  }
  ss[threadIdx.x] = s; ss2[threadIdx.x] = s2;
  __syncthreads();
  for (int off = 128; off > 0; off >>= 1) {
    if (threadIdx.x < off) {
      ss[threadIdx.x]  += ss[threadIdx.x + off];
      ss2[threadIdx.x] += ss2[threadIdx.x + off];
    }
    __syncthreads();
  }
  if (threadIdx.x == 0) {
    const float inv = 1.0f / (float)(CPG * HW);
    float m   = ss[0] * inv;
    float var = ss2[0] * inv - m * m;
    mu[bg] = m;
    rs[bg] = rsqrtf(var + 1e-5f);
  }
}

// ------------------------------------------------------------------
// GroupNorm pass 2: normalize + affine, emit bf16 pairs (dword stores)
// in native (b,c,p) layout == column-major A for the projection GEMMs.
// ------------------------------------------------------------------
__global__ void gn_norm_kernel(const float* __restrict__ x,
                               const float* __restrict__ mu, const float* __restrict__ rs,
                               const float* __restrict__ gamma, const float* __restrict__ beta,
                               unsigned* __restrict__ xn2) {
  size_t i2  = (size_t)blockIdx.x * blockDim.x + threadIdx.x;  // pair index
  size_t idx = i2 * 2;
  int b   = (int)(idx >> 21);              // CCH*HW = 2^21
  int rem = (int)(idx & ((1u << 21) - 1));
  int c   = rem >> 12;                     // HW = 2^12 (pairs never cross channels)
  int bg  = b * GROUPS + (c >> 4);
  float g = rs[bg] * gamma[c];
  float m = mu[bg], bb = beta[c];
  float v0 = (x[idx]     - m) * g + bb;
  float v1 = (x[idx + 1] - m) * g + bb;
  xn2[i2] = (unsigned)f2bf(v0) | ((unsigned)f2bf(v1) << 16);
}

// ------------------------------------------------------------------
// Pack fp32 weight (R x C) into bf16 transposed (C x R) for Bt operand.
// ------------------------------------------------------------------
__global__ void packT_kernel(const float* __restrict__ in, unsigned short* __restrict__ out,
                             int R, int Ccol) {
  int idx = blockIdx.x * 256 + threadIdx.x;     // grid covers exactly R*Ccol
  int r = idx / Ccol, c = idx % Ccol;
  out[(size_t)c * R + r] = f2bf(in[idx]);
}

// ------------------------------------------------------------------
// bf16 NT GEMM via v_wmma_f32_16x16x32_bf16.
//   A  : column-major, storage (K rows, M cols), lda = M-dim stride
//   Bt : row-major,    storage (N rows, K cols), ldb = K-dim stride
//   C(m,n) = alpha * ( sum_k A[k][m]*Bt[n][k] + bias[n] )
// MODE 0: bf16 store C[m*ldc+n]
// MODE 1: bf16 store C[n*ldc+m]  (transposed; packed b128 stores)
// MODE 2: f32  store C[n*ldc+m] = resid[..] + val  (fused residual, b128)
// 256 threads = 8 waves; 128x128 block tile; wave tile 32x64 = 2x4 frags
// -> 8 WMMA per 12 fragment b128 loads per wave per K-step.
// Staging split: waves 0-3 transpose-stage A (b64 packed), waves 4-7 copy B.
// 2-stage pipeline: next global tile prefetched into regs during math.
// Dims are multiples of 128 (M,N) and 32 (K) -> EXEC all-ones for WMMA.
// ------------------------------------------------------------------
template <int MODE, bool HB>
__global__ __launch_bounds__(256) void gemm_nt_wmma(
    const unsigned short* __restrict__ A,  int lda, long long strideA,
    const unsigned short* __restrict__ Bt, int ldb, long long strideB,
    const float* __restrict__ bias,
    void* __restrict__ Cptr, int ldc, long long strideC,
    const float* __restrict__ resid, long long strideR,
    int K, float alpha) {

  __shared__ unsigned short lA[128 * 40];  // padded stride 40 halves (80B rows)
  __shared__ unsigned short lB[128 * 40];

  const int m0 = blockIdx.x * 128;
  const int n0 = blockIdx.y * 128;
  const int b  = blockIdx.z;
  A  += (long long)b * strideA;
  Bt += (long long)b * strideB;

  const int tid  = threadIdx.x;
  const int wave = tid >> 5;
  const int lane = tid & 31;
  const int wm = (wave & 3) * 32;          // 4 waves along M
  const int wn = (wave >> 2) * 64;         // 2 waves along N
  const int r  = lane & 15;
  const int h  = lane >> 4;

  const bool isA = (tid < 128);            // waves 0-3: A stager; 4-7: B stager
  const int  t   = isA ? tid : tid - 128;
  const int  kq  = t >> 4;                 // A: 4 k-rows starting at 4*kq (0..7)
  const int  mm  = (t & 15) * 8;           // A: 8 m values (0..120)

  uv4 r0, r1, r2, r3;
  auto loadStage = [&](int k0) {
    if (isA) {
      const unsigned short* p = A + (size_t)(k0 + 4 * kq) * lda + m0 + mm;
      r0 = *(const uv4*)p;
      r1 = *(const uv4*)(p + lda);
      r2 = *(const uv4*)(p + 2 * lda);
      r3 = *(const uv4*)(p + 3 * lda);
    } else {
      const unsigned short* p = Bt + (size_t)(n0 + t) * ldb + k0;
      r0 = *(const uv4*)p;
      r1 = *(const uv4*)(p + 8);
      r2 = *(const uv4*)(p + 16);
      r3 = *(const uv4*)(p + 24);
    }
  };
  auto storeStage = [&]() {
    if (isA) {
      // register transpose: (k,m) -> lA[m][k], b64 stores covering k=4kq..4kq+3
#pragma unroll
      for (int j = 0; j < 4; ++j) {
        unsigned a0 = r0[j], a1 = r1[j], a2 = r2[j], a3 = r3[j];
        uv2 dlo = { (a0 & 0xffffu) | (a1 << 16), (a2 & 0xffffu) | (a3 << 16) };
        uv2 dhi = { (a0 >> 16) | (a1 & 0xffff0000u), (a2 >> 16) | (a3 & 0xffff0000u) };
        *(uv2*)&lA[(mm + 2 * j)     * 40 + 4 * kq] = dlo;
        *(uv2*)&lA[(mm + 2 * j + 1) * 40 + 4 * kq] = dhi;
      }
    } else {
      unsigned short* q = &lB[t * 40];
      *(uv4*)(q)      = r0;
      *(uv4*)(q + 8)  = r1;
      *(uv4*)(q + 16) = r2;
      *(uv4*)(q + 24) = r3;
    }
  };

  v8f acc[2][4];
#pragma unroll
  for (int i = 0; i < 2; ++i)
#pragma unroll
    for (int j = 0; j < 4; ++j) acc[i][j] = (v8f){};

  union FU { uv4 u[2]; v16bf v; };

  const int ksteps = K >> 5;
  loadStage(0);
  for (int kb = 0; kb < ksteps; ++kb) {
    storeStage();
    __syncthreads();
    if (kb + 1 < ksteps) loadStage((kb + 1) << 5);   // prefetch next tile

    FU a[2], bb[4];
#pragma unroll
    for (int i = 0; i < 2; ++i) {
      const unsigned short* p = &lA[(wm + i * 16 + r) * 40 + h * 16];
      a[i].u[0] = *(const uv4*)p; a[i].u[1] = *(const uv4*)(p + 8);
    }
#pragma unroll
    for (int j = 0; j < 4; ++j) {
      const unsigned short* p = &lB[(wn + j * 16 + r) * 40 + h * 16];
      bb[j].u[0] = *(const uv4*)p; bb[j].u[1] = *(const uv4*)(p + 8);
    }
#pragma unroll
    for (int i = 0; i < 2; ++i)
#pragma unroll
      for (int j = 0; j < 4; ++j)
        acc[i][j] = __builtin_amdgcn_wmma_f32_16x16x32_bf16(
            false, a[i].v, false, bb[j].v, (short)0, acc[i][j], false, false);

    __syncthreads();
  }

  // --- epilogue: C/D layout VGPR e -> row (h?8:0)+e, col r; per lane the 8
  //     m's are consecutive -> packed b128 stores in transposed modes. ---
#pragma unroll
  for (int i = 0; i < 2; ++i) {
#pragma unroll
    for (int j = 0; j < 4; ++j) {
      v8f av = acc[i][j];
      const int mbase = m0 + wm + i * 16 + h * 8;
      const int nn    = n0 + wn + j * 16 + r;
      const float bia = HB ? bias[nn] : 0.0f;

      if (MODE == 0) {
        unsigned short* Cb = (unsigned short*)Cptr + (long long)b * strideC;
#pragma unroll
        for (int e = 0; e < 8; ++e)
          Cb[(size_t)(mbase + e) * ldc + nn] = f2bf((av[e] + bia) * alpha);
      } else if (MODE == 1) {
        unsigned short* Cb = (unsigned short*)Cptr + (long long)b * strideC
                             + (size_t)nn * ldc + mbase;
        uv4 d;
#pragma unroll
        for (int e2 = 0; e2 < 4; ++e2) {
          unsigned lo = f2bf((av[2 * e2]     + bia) * alpha);
          unsigned hi = f2bf((av[2 * e2 + 1] + bia) * alpha);
          d[e2] = lo | (hi << 16);
        }
        *(uv4*)Cb = d;
      } else {
        float*       Cb = (float*)Cptr + (long long)b * strideC + (size_t)nn * ldc + mbase;
        const float* Rb = resid        + (long long)b * strideR + (size_t)nn * ldc + mbase;
        v4f rv0 = *(const v4f*)Rb;
        v4f rv1 = *(const v4f*)(Rb + 4);
        v4f o0, o1;
#pragma unroll
        for (int e = 0; e < 4; ++e) {
          o0[e] = rv0[e] + (av[e]     + bia) * alpha;
          o1[e] = rv1[e] + (av[e + 4] + bia) * alpha;
        }
        *(v4f*)Cb       = o0;
        *(v4f*)(Cb + 4) = o1;
      }
    }
  }
}

// ------------------------------------------------------------------
// Host orchestration
// ------------------------------------------------------------------
extern "C" void kernel_launch(void* const* d_in, const int* in_sizes, int n_in,
                              void* d_out, int out_size, void* d_ws, size_t ws_size,
                              hipStream_t stream) {
  (void)in_sizes; (void)n_in; (void)out_size; (void)ws_size;

  const float* x     = (const float*)d_in[0];
  const float* gamma = (const float*)d_in[1];
  const float* beta  = (const float*)d_in[2];
  const float* Wq    = (const float*)d_in[3];
  const float* bq    = (const float*)d_in[4];
  const float* Wk    = (const float*)d_in[5];
  const float* bk    = (const float*)d_in[6];
  const float* Wv    = (const float*)d_in[7];
  const float* bv    = (const float*)d_in[8];
  const float* orth  = (const float*)d_in[9];
  float* out = (float*)d_out;

  // workspace carve-up (256B aligned)
  char* ws = (char*)d_ws;
  size_t o = 0;
  auto alloc = [&](size_t bytes) -> char* {
    char* p = ws + o; o += (bytes + 255) & ~(size_t)255; return p;
  };
  float* mu            = (float*)alloc(512 * 4);
  float* rs            = (float*)alloc(512 * 4);
  unsigned short* WqT  = (unsigned short*)alloc((size_t)NFD * CCH * 2);
  unsigned short* WkT  = (unsigned short*)alloc((size_t)NFD * CCH * 2);
  unsigned short* WvT  = (unsigned short*)alloc((size_t)CCH * CCH * 2);
  unsigned short* orT  = (unsigned short*)alloc((size_t)NFD * NFD * 2);
  unsigned short* xn   = (unsigned short*)alloc((size_t)NB * CCH * HW * 2);  // 64MB
  unsigned short* q0T  = (unsigned short*)alloc((size_t)NB * NFD * HW * 2);  // 32MB
  unsigned short* k0T  = (unsigned short*)alloc((size_t)NB * NFD * HW * 2);  // 32MB
  unsigned short* vT   = (unsigned short*)alloc((size_t)NB * CCH * HW * 2);  // 64MB
  // reuse: xn dead after projections; q0T dead after q-orth GEMM
  unsigned short* q1T  = xn;                               // (b, 256, 4096)
  unsigned short* k1   = xn + (size_t)NB * NFD * HW;       // (b, 4096, 256)
  unsigned short* kvT  = q0T;                              // (b, 512, 256)

  const long long sXN = (long long)CCH * HW;   // 2097152
  const long long sQ  = (long long)NFD * HW;   // 1048576
  const long long sKV = (long long)CCH * NFD;  // 131072
  const float scale = 1.0f / (16.0f + 1e-6f);  // 1/(sqrt(256)+1e-6)

  gn_stats_kernel<<<NB * GROUPS, 256, 0, stream>>>(x, mu, rs);
  gn_norm_kernel<<<(NB * CCH * HW / 2) / 256, 256, 0, stream>>>(
      x, mu, rs, gamma, beta, (unsigned*)xn);
  packT_kernel<<<(CCH * NFD) / 256, 256, 0, stream>>>(Wq, WqT, CCH, NFD);
  packT_kernel<<<(CCH * NFD) / 256, 256, 0, stream>>>(Wk, WkT, CCH, NFD);
  packT_kernel<<<(CCH * CCH) / 256, 256, 0, stream>>>(Wv, WvT, CCH, CCH);
  packT_kernel<<<(NFD * NFD) / 256, 256, 0, stream>>>(orth, orT, NFD, NFD);

  // q0^T = (xf @ Wq + bq)^T        M=4096 N=256 K=512
  gemm_nt_wmma<1, true><<<dim3(32, 2, NB), 256, 0, stream>>>(xn, HW, sXN, WqT, CCH, 0, bq,
      q0T, HW, sQ, nullptr, 0, CCH, 1.0f);
  // k0^T = (xf @ Wk + bk)^T
  gemm_nt_wmma<1, true><<<dim3(32, 2, NB), 256, 0, stream>>>(xn, HW, sXN, WkT, CCH, 0, bk,
      k0T, HW, sQ, nullptr, 0, CCH, 1.0f);
  // v^T  = (xf @ Wv + bv)^T        M=4096 N=512 K=512
  gemm_nt_wmma<1, true><<<dim3(32, 4, NB), 256, 0, stream>>>(xn, HW, sXN, WvT, CCH, 0, bv,
      vT, HW, sXN, nullptr, 0, CCH, 1.0f);
  // q1^T = ((q0 @ orth) * scale)^T M=4096 N=256 K=256
  gemm_nt_wmma<1, false><<<dim3(32, 2, NB), 256, 0, stream>>>(q0T, HW, sQ, orT, NFD, 0, nullptr,
      q1T, HW, sQ, nullptr, 0, NFD, scale);
  // k1   = (k0 @ orth) * scale     (row-major = col-major A for kv GEMM)
  gemm_nt_wmma<0, false><<<dim3(32, 2, NB), 256, 0, stream>>>(k0T, HW, sQ, orT, NFD, 0, nullptr,
      k1, NFD, sQ, nullptr, 0, NFD, scale);
  // kv^T = (k1^T @ v)^T            M=256 N=512 K=4096
  gemm_nt_wmma<1, false><<<dim3(2, 4, NB), 256, 0, stream>>>(k1, NFD, sQ, vT, HW, sXN, nullptr,
      kvT, NFD, sKV, nullptr, 0, HW, 1.0f);
  // out = x + (q1 @ kv)^T          M=4096 N=512 K=256, f32 + fused residual
  gemm_nt_wmma<2, false><<<dim3(32, 4, NB), 256, 0, stream>>>(q1T, HW, sQ, kvT, NFD, sKV, nullptr,
      out, HW, sXN, x, sXN, NFD, 1.0f);
}
